// DWDeformConv_30331059044942
// MI455X (gfx1250) — matrix operand order repaired and checked by reading, not compiled, not run on hardware
//
#include <hip/hip_runtime.h>
#include <hip/hip_bf16.h>
#include <math.h>

// ---------------- problem constants ----------------
#define NB   4
#define C1C  128
#define C2C  36
#define C2P  48          // C2 padded to 3 WMMA N-tiles
#define KK   3
#define HH   192
#define WWI  192
#define HOC  190
#define WOC  190
#define DGC  2
#define GRP  4
#define TILE_M 128
#define EPSBN 1e-5f

typedef _Float16 half_t;
typedef __attribute__((ext_vector_type(16))) _Float16 v16h;
typedef __attribute__((ext_vector_type(8)))  _Float16 v8h;
typedef __attribute__((ext_vector_type(8)))  float    v8f;

union F16x16 { v16h v; v8h h[2]; };

__device__ __forceinline__ float mish_f(float v) {
    float sp = (v > 20.f) ? v : log1pf(expf(v));
    return v * tanhf(sp);
}

// ---------------------------------------------------------------------------
// Kernel 1: grouped offset conv (VALID, 3x3, groups=4), offsets -> d_ws
// ---------------------------------------------------------------------------
__global__ __launch_bounds__(256)
void offset_conv_kernel(const float* __restrict__ x,
                        const float* __restrict__ cw,
                        float* __restrict__ offs) {
    __shared__ float w[C2C * 32 * 9];   // 41472 floats? no: 36*32*9 = 10368 floats = 41.5KB
    for (int i = threadIdx.x; i < C2C * 32 * 9; i += 256) w[i] = cw[i];
    __syncthreads();

    long gid = (long)blockIdx.x * 256 + threadIdx.x;
    const long total = (long)NB * HOC * WOC;
    if (gid >= total) return;

    int n   = (int)(gid / (HOC * WOC));
    int pix = (int)(gid % (HOC * WOC));
    int ho  = pix / WOC;
    int wo  = pix - ho * WOC;

    float acc[C2C];
#pragma unroll
    for (int c = 0; c < C2C; ++c) acc[c] = 0.f;

#pragma unroll
    for (int g4 = 0; g4 < GRP; ++g4) {
        for (int ci = 0; ci < 32; ++ci) {
            const float* p = x + (((size_t)(n * C1C + g4 * 32 + ci) * HH + ho) * WWI + wo);
            float v[9];
            v[0] = p[0];        v[1] = p[1];        v[2] = p[2];
            v[3] = p[WWI];      v[4] = p[WWI + 1];  v[5] = p[WWI + 2];
            v[6] = p[2 * WWI];  v[7] = p[2 * WWI + 1]; v[8] = p[2 * WWI + 2];
#pragma unroll
            for (int oc = 0; oc < 9; ++oc) {
                const float* ww = &w[(((g4 * 9 + oc) * 32) + ci) * 9];
                float s = 0.f;
#pragma unroll
                for (int q = 0; q < 9; ++q) s = fmaf(v[q], ww[q], s);
                acc[g4 * 9 + oc] += s;
            }
        }
    }

#pragma unroll
    for (int c2 = 0; c2 < C2C; ++c2)
        offs[(((size_t)n * C2C + c2) * HOC + ho) * WOC + wo] = acc[c2];
}

// ---------------------------------------------------------------------------
// Kernel 2: repack deform_w [36][128][3][3] f32 -> wB [9][48][128] f16
// ---------------------------------------------------------------------------
__global__ __launch_bounds__(256)
void prep_w_kernel(const float* __restrict__ dw, half_t* __restrict__ wB) {
    int idx = blockIdx.x * 256 + threadIdx.x;
    if (idx >= 9 * C2P * C1C) return;
    int c1   = idx & (C1C - 1);
    int rest = idx >> 7;
    int c2p  = rest % C2P;
    int t    = rest / C2P;
    float v = 0.f;
    if (c2p < C2C) v = dw[((size_t)c2p * C1C + c1) * 9 + t];
    wB[((size_t)t * C2P + c2p) * C1C + c1] = (half_t)v;
}

// ---------------------------------------------------------------------------
// Kernel 3: deformable sampling + WMMA GEMM + BN + Mish
//   block = 256 threads (8 waves), one block = 128 output pixels of batch n
// ---------------------------------------------------------------------------
#define AS_STRIDE 136   // 128 + 8 halfs padding -> conflict-free A-frag reads

__global__ __launch_bounds__(256)
void deform_gemm_kernel(const float* __restrict__ x,
                        const float* __restrict__ offs,
                        const half_t* __restrict__ wB,
                        const float* __restrict__ bn_g,
                        const float* __restrict__ bn_b,
                        const float* __restrict__ bn_m,
                        const float* __restrict__ bn_v,
                        float* __restrict__ out) {
    __shared__ half_t As[TILE_M][AS_STRIDE];     // 34816 B: sampled tap tile f16
    __shared__ half_t Bs[C2P][C1C];              // 12288 B: weights, [col][k]
    __shared__ int    cIdx[TILE_M][DGC][4];      // 4096 B: clamped corner offsets
    __shared__ float  cW[TILE_M][DGC][4];        // 4096 B: masked bilinear weights
    __shared__ float  bnScale[C2C], bnShift[C2C];

    const int tid    = threadIdx.x;
    const int n      = blockIdx.y;
    const int tile0  = blockIdx.x * TILE_M;
    const int lane   = tid & 31;
    const int wave   = tid >> 5;
    const int lane15 = lane & 15;
    const int khalf  = lane >> 4;
    const int m0     = wave * 16;

    if (tid < C2C) {
        float inv = bn_g[tid] * rsqrtf(bn_v[tid] + EPSBN);
        bnScale[tid] = inv;
        bnShift[tid] = bn_b[tid] - bn_m[tid] * inv;
    }

    v8f acc[3] = {};

    const int smp_m  = tid >> 1;      // pixel within tile for sampling phases
    const int smp_dg = tid & 1;       // deformable group

    for (int t = 0; t < 9; ++t) {
        const int ki = t / 3, kj = t % 3;

        // ---- stage B tile for this tap (f16, [c2p][c1], K-contiguous) ----
        {
            const unsigned* src = (const unsigned*)(wB + (size_t)t * C2P * C1C);
            unsigned* dst = (unsigned*)&Bs[0][0];
            for (int i = tid; i < (C2P * C1C) / 2; i += 256) dst[i] = src[i];
        }

        // ---- per-(pixel, dg) bilinear coordinate precompute ----
        {
            float w4[4] = {0.f, 0.f, 0.f, 0.f};
            int   i4[4] = {0, 0, 0, 0};
            int pix = tile0 + smp_m;
            if (pix < HOC * WOC) {
                int ho = pix / WOC, wo = pix - ho * WOC;
                const float* ob = offs +
                    (((size_t)(n * C2C + smp_dg * 18 + t * 2) * HOC + ho) * WOC + wo);
                float py = (float)(ho + ki) + ob[0];
                float px = (float)(wo + kj) + ob[(size_t)HOC * WOC];
                float y0f = floorf(py), x0f = floorf(px);
                float fy = py - y0f, fx = px - x0f;
                int iy0 = (int)y0f, ix0 = (int)x0f;
                int iy1 = iy0 + 1,  ix1 = ix0 + 1;
                float vy0 = (iy0 >= 0 && iy0 < HH) ? 1.f : 0.f;
                float vy1 = (iy1 >= 0 && iy1 < HH) ? 1.f : 0.f;
                float vx0 = (ix0 >= 0 && ix0 < WWI) ? 1.f : 0.f;
                float vx1 = (ix1 >= 0 && ix1 < WWI) ? 1.f : 0.f;
                int iy0c = min(max(iy0, 0), HH - 1);
                int iy1c = min(max(iy1, 0), HH - 1);
                int ix0c = min(max(ix0, 0), WWI - 1);
                int ix1c = min(max(ix1, 0), WWI - 1);
                w4[0] = (1.f - fy) * (1.f - fx) * vy0 * vx0;
                w4[1] = (1.f - fy) * fx         * vy0 * vx1;
                w4[2] = fy * (1.f - fx)         * vy1 * vx0;
                w4[3] = fy * fx                 * vy1 * vx1;
                i4[0] = iy0c * WWI + ix0c;
                i4[1] = iy0c * WWI + ix1c;
                i4[2] = iy1c * WWI + ix0c;
                i4[3] = iy1c * WWI + ix1c;
            }
#pragma unroll
            for (int q = 0; q < 4; ++q) {
                cIdx[smp_m][smp_dg][q] = i4[q];
                cW[smp_m][smp_dg][q]   = w4[q];
            }
        }
        __syncthreads();

        // ---- gather-sample 64 channels per thread into the f16 A tile ----
        {
            const int   i00 = cIdx[smp_m][smp_dg][0];
            const int   i01 = cIdx[smp_m][smp_dg][1];
            const int   i10 = cIdx[smp_m][smp_dg][2];
            const int   i11 = cIdx[smp_m][smp_dg][3];
            const float w00 = cW[smp_m][smp_dg][0];
            const float w01 = cW[smp_m][smp_dg][1];
            const float w10 = cW[smp_m][smp_dg][2];
            const float w11 = cW[smp_m][smp_dg][3];
            const bool any = (w00 != 0.f) | (w01 != 0.f) | (w10 != 0.f) | (w11 != 0.f);
            const float* p = x + ((size_t)(n * C1C + smp_dg * 64) * HH * WWI);
            half_t* arow = &As[smp_m][smp_dg * 64];
            for (int cc = 0; cc < 64; ++cc) {
                float v = 0.f;
                if (any) {
                    const float* pc = p + (size_t)cc * (HH * WWI);
                    v = fmaf(w00, pc[i00],
                        fmaf(w01, pc[i01],
                        fmaf(w10, pc[i10], w11 * pc[i11])));
                }
                arow[cc] = (half_t)v;
            }
        }
        __syncthreads();

        // ---- WMMA: 16(M) x 48(N) x 128(K) per wave for this tap ----
#pragma unroll
        for (int kc = 0; kc < 4; ++kc) {
            F16x16 a;
            const half_t* ar = &As[m0 + lane15][kc * 32 + khalf * 8];
            a.h[0] = *(const v8h*)ar;          // K = kc*32 + khalf*8 + [0..7]
            a.h[1] = *(const v8h*)(ar + 16);   // K = kc*32 + 16 + khalf*8 + [0..7]
#pragma unroll
            for (int nt = 0; nt < 3; ++nt) {
                F16x16 bf;
                const half_t* br = &Bs[nt * 16 + lane15][kc * 32 + khalf * 8];
                bf.h[0] = *(const v8h*)br;     // K = kc*32 + khalf*8 + [0..7]
                bf.h[1] = *(const v8h*)(br + 16);
                acc[nt] = __builtin_amdgcn_wmma_f32_16x16x32_f16(
                    false, a.v, false, bf.v, (short)0, acc[nt], false, false);
            }
        }
        __syncthreads();
    }

    // ---- epilogue: BN + Mish, scatter to NCHW ----
#pragma unroll
    for (int nt = 0; nt < 3; ++nt) {
        const int c2 = nt * 16 + lane15;
        if (c2 >= C2C) continue;
        const float sc = bnScale[c2], sh = bnShift[c2];
#pragma unroll
        for (int r = 0; r < 8; ++r) {
            int m   = m0 + r + khalf * 8;      // D layout: VGPR r -> M=r / M=8+r
            int pix = tile0 + m;
            if (pix >= HOC * WOC) continue;
            int ho = pix / WOC, wo = pix - ho * WOC;
            float v = fmaf(acc[nt][r], sc, sh);
            out[(((size_t)n * C2C + c2) * HOC + ho) * WOC + wo] = mish_f(v);
        }
    }
}

// ---------------------------------------------------------------------------
// launch
// ---------------------------------------------------------------------------
extern "C" void kernel_launch(void* const* d_in, const int* in_sizes, int n_in,
                              void* d_out, int out_size, void* d_ws, size_t ws_size,
                              hipStream_t stream) {
    const float* x        = (const float*)d_in[0];
    const float* conv_w   = (const float*)d_in[1];
    const float* deform_w = (const float*)d_in[2];
    const float* bn_g     = (const float*)d_in[3];
    const float* bn_b     = (const float*)d_in[4];
    const float* bn_m     = (const float*)d_in[5];
    const float* bn_v     = (const float*)d_in[6];
    float* out = (float*)d_out;

    const size_t offs_elems = (size_t)NB * C2C * HOC * WOC;          // 5,198,400 f32
    float*  offs = (float*)d_ws;
    half_t* wB   = (half_t*)((char*)d_ws + offs_elems * sizeof(float)); // +110,592 B

    const long total1 = (long)NB * HOC * WOC;
    offset_conv_kernel<<<dim3((unsigned)((total1 + 255) / 256)), dim3(256), 0, stream>>>(
        x, conv_w, offs);

    prep_w_kernel<<<dim3((9 * C2P * C1C + 255) / 256), dim3(256), 0, stream>>>(
        deform_w, wB);

    const int tiles = (HOC * WOC + TILE_M - 1) / TILE_M;             // 283
    deform_gemm_kernel<<<dim3(tiles, NB), dim3(256), 0, stream>>>(
        x, offs, wB, bn_g, bn_b, bn_m, bn_v, out);
}
